// SegerZoomCropCat_55508157334083
// MI455X (gfx1250) — matrix operand order repaired and checked by reading, not compiled, not run on hardware
//
#include <hip/hip_runtime.h>
#include <hip/hip_bf16.h>
#include <math.h>

// ---------------------------------------------------------------------------
// SegerZoomCropCat for MI455X (gfx1250, wave32, WMMA)
// Pipeline:
//   repack weights -> build alias pyramid (x_as + f16 NHWC32) -> seg0 CNN
//   (WMMA conv3x3 x2) -> dmap head -> separable 17-tap blur -> 5x5 grid ->
//   grid-sample (x_gs + f16 NHWC32) -> seg CNN (WMMA conv3x3 x4) -> final head
// ---------------------------------------------------------------------------

typedef __attribute__((ext_vector_type(16))) _Float16 v16h;
typedef __attribute__((ext_vector_type(8)))  _Float16 v8h;
typedef __attribute__((ext_vector_type(8)))  float    v8f;

#define Bn   8
#define Cc   4
#define Hh   1024
#define Ww   1024
#define HS   128
#define WS   128
#define Aa   4
#define AREA 20
#define NPIX (Bn*HS*WS)          // 131072

__constant__ float c_mean[5] = {0.485f, 0.456f, 0.406f, 0.5f, 0.5f};
__constant__ float c_std [5] = {0.229f, 0.224f, 0.225f, 0.25f, 0.25f};

// ---------------------------------------------------------------------------
// Weight repack: OIHW f32 -> CDNA5 WMMA B-fragment layout, f16.
// Packed layout: [kc(9)][ntile(2)][lane(32)][elem(16)]
//   lane<16 : elem i -> (K=i,    N=lane)
//   lane>=16: elem i -> (K=16+i, N=lane-16)
//   K (within chunk) = cin (zero-padded to 32), chunk kc = ky*3+kx
// ---------------------------------------------------------------------------
__global__ void repack_w(const float* __restrict__ W, _Float16* __restrict__ dst,
                         int Cin, int Cout) {
  int t = blockIdx.x * blockDim.x + threadIdx.x;
  if (t >= 9*2*32*16) return;
  int i     =  t        & 15;
  int lane  = (t >> 4)  & 31;
  int ntile = (t >> 9)  & 1;
  int kc    =  t >> 10;           // 0..8
  int ky = kc / 3, kx = kc % 3;
  int n = ntile * 16 + (lane & 15);
  int k = ((lane >> 4) << 4) + i; // cin
  float v = 0.0f;
  if (n < Cout && k < Cin)
    v = W[((n * Cin + k) * 3 + ky) * 3 + kx];
  dst[t] = (_Float16)v;
}

__global__ void pad_bias(const float* __restrict__ b, float* __restrict__ dst, int Cout) {
  int t = threadIdx.x;
  if (t < 32) dst[t] = (t < Cout) ? b[t] : 0.0f;
}

// ---------------------------------------------------------------------------
// Alias pyramid: per output pixel compute avg-pooled crops (4 ch) + analytic
// max-pooled normalized distance, write x_as (f32, d_out) and normalized
// f16 NHWC32 activations for seg0.
// ---------------------------------------------------------------------------
__global__ __launch_bounds__(256) void build_xas(
    const float* __restrict__ x, const float* __restrict__ xy,
    float* __restrict__ xas_out, _Float16* __restrict__ xn) {
  int t = blockIdx.x * blockDim.x + threadIdx.x;
  if (t >= NPIX) return;
  int pw = t & 127, ph = (t >> 7) & 127, b = t >> 14;

  float hidx = xy[b*2+0] * (float)(Hh-1);
  float widx = xy[b*2+1] * (float)(Ww-1);
  int hI = (int)hidx, wI = (int)widx;
  float mh = fmaxf(hidx*hidx, (1023.f-hidx)*(1023.f-hidx));
  float mw = fmaxf(widx*widx, (1023.f-widx)*(1023.f-widx));
  float inv_maxd = rsqrtf(mh + mw);

  const float* xb  = x + (size_t)b * Cc * Hh * Ww;
  _Float16*    xnp = xn + (size_t)t * 32;
  float*       xop = xas_out + (size_t)b * Aa * 5 * HS * WS;

  #pragma unroll
  for (int a = 0; a < 4; ++a) {
    const int chs = Hh >> a;
    const int f   = 8 >> a;
    int ua = hI - chs/2; ua = ua < 0 ? 0 : (ua > Hh-chs ? Hh-chs : ua);
    int la = wI - chs/2; la = la < 0 ? 0 : (la > Ww-chs ? Ww-chs : la);
    int y0 = ua + ph * f, x0 = la + pw * f;
    float inv_area = 1.0f / (float)(f * f);
    #pragma unroll
    for (int c = 0; c < 4; ++c) {
      float s = 0.0f;
      const float* p = xb + (size_t)c * Hh * Ww + (size_t)y0 * Ww + x0;
      for (int i = 0; i < f; ++i)
        for (int j = 0; j < f; ++j) s += p[i * Ww + j];
      float v = s * inv_area;
      xop[(a*5 + c) * (HS*WS) + ph*WS + pw] = v;
      xnp[a*5 + c] = (_Float16)((v - c_mean[c]) / c_std[c]);
    }
    // max-pooled normalized distance: max of sqrt over block = corner of block
    float dh0 = (float)y0 - hidx,        dh1 = (float)(y0 + f - 1) - hidx;
    float dw0 = (float)x0 - widx,        dw1 = (float)(x0 + f - 1) - widx;
    float d = sqrtf(fmaxf(dh0*dh0, dh1*dh1) + fmaxf(dw0*dw0, dw1*dw1)) * inv_maxd;
    xop[(a*5 + 4) * (HS*WS) + ph*WS + pw] = d;
    xnp[a*5 + 4] = (_Float16)((d - c_mean[4]) / c_std[4]);
  }
  #pragma unroll
  for (int c = 20; c < 32; ++c) xnp[c] = (_Float16)0.0f;
}

// ---------------------------------------------------------------------------
// Implicit-GEMM conv3x3 (SAME, zero pad) with V_WMMA_F32_16X16X32_F16.
// Input / output: f16 NHWC, 32 channels (zero padded). 128 threads = 4 waves;
// each wave computes 16 pixels x 32 output channels (2 WMMA N-tiles) via
// 9 K-chunks of 32 (one per 3x3 tap). Input row tile staged in LDS.
// ---------------------------------------------------------------------------
__global__ __launch_bounds__(128) void conv3x3_wmma(
    const _Float16* __restrict__ in,   // B*128*128*32
    const _Float16* __restrict__ wpk,  // 9*2*32*16 packed B fragments
    const float*    __restrict__ bias, // 32 (zero padded)
    _Float16*       __restrict__ out,  // B*128*128*32
    int relu) {
  __shared__ _Float16 tile[3 * 66 * 32];   // rows y-1..y+1, cols x0-1..x0+64

  const int lane = threadIdx.x & 31;
  const int wid  = threadIdx.x >> 5;       // wave 0..3 -> 16-pixel segment
  const int blk  = blockIdx.x;
  const int xseg = blk & 1;
  const int y    = (blk >> 1) & 127;
  const int b    = blk >> 8;
  const int x0   = xseg * 64;

  __builtin_prefetch(wpk, 0, 3);           // global_prefetch_b8 the weights

  const _Float16* inb = in + (size_t)b * (HS * WS * 32);

  v8h zero = {};
  for (int idx = threadIdx.x; idx < 3 * 66; idx += 128) {
    int r = idx / 66, col = idx % 66;
    int iy = y + r - 1, ix = x0 + col - 1;
    _Float16* dst = &tile[(r * 66 + col) * 32];
    if (iy >= 0 && iy < HS && ix >= 0 && ix < WS) {
      const _Float16* src = inb + ((size_t)iy * WS + ix) * 32;
      #pragma unroll
      for (int q = 0; q < 4; ++q) *(v8h*)&dst[q*8] = *(const v8h*)&src[q*8];
    } else {
      #pragma unroll
      for (int q = 0; q < 4; ++q) *(v8h*)&dst[q*8] = zero;
    }
  }
  __syncthreads();

  const int laneM = lane & 15;
  const int hi    = lane >> 4;             // half-wave select

  v8f acc0 = {}; v8f acc1 = {};
  #pragma unroll
  for (int kc = 0; kc < 9; ++kc) {
    const int ky = kc / 3, kx = kc % 3;
    const int col = wid * 16 + laneM + kx; // tile col (x-1 origin built in)
    const _Float16* base = &tile[((size_t)ky * 66 + col) * 32];
    // A fragment (contiguous K, matching packed B):
    //   lane<16 : elems 0..15 -> K 0..15
    //   lane>=16: elems 0..15 -> K 16..31
    v8h lo  = *(const v8h*)&base[hi * 16];
    v8h hi8 = *(const v8h*)&base[hi * 16 + 8];
    v16h a;
    #pragma unroll
    for (int i = 0; i < 8; ++i) { a[i] = lo[i]; a[i + 8] = hi8[i]; }

    const _Float16* wb = wpk + ((size_t)(kc * 2) * 32 + lane) * 16;
    v16h b0 = *(const v16h*)wb;            // N tile 0..15
    v16h b1 = *(const v16h*)(wb + 32*16);  // N tile 16..31

    acc0 = __builtin_amdgcn_wmma_f32_16x16x32_f16(false, a, false, b0,
                                                  (short)0, acc0, false, false);
    acc1 = __builtin_amdgcn_wmma_f32_16x16x32_f16(false, a, false, b1,
                                                  (short)0, acc1, false, false);
  }

  // C/D layout: VGPR r, lane<16 -> (M=r, N=lane); lane>=16 -> (M=8+r, N=lane-16)
  const int  N  = laneM;
  const float b0f = bias[N], b1f = bias[16 + N];
  _Float16* outb = out + (size_t)b * (HS * WS * 32) + (size_t)y * WS * 32;
  #pragma unroll
  for (int r = 0; r < 8; ++r) {
    int M  = hi * 8 + r;
    int px = x0 + wid * 16 + M;
    float v0 = acc0[r] + b0f;
    float v1 = acc1[r] + b1f;
    if (relu) { v0 = v0 > 0.f ? v0 : 0.f; v1 = v1 > 0.f ? v1 : 0.f; }
    outb[(size_t)px * 32 + N]      = (_Float16)v0;
    outb[(size_t)px * 32 + 16 + N] = (_Float16)v1;
  }
}

// ---------------------------------------------------------------------------
// seg0 head: 1x1 conv (4->1) + priori + sigmoid, scaled by unnormalized
// saliency channel x_as[:,0,3] -> dmap
// ---------------------------------------------------------------------------
__global__ void dmap_head(const _Float16* __restrict__ feat,
                          const float* __restrict__ w, const float* __restrict__ bb,
                          const float* __restrict__ xy, const float* __restrict__ xas,
                          float* __restrict__ dmap) {
  int t = blockIdx.x * blockDim.x + threadIdx.x;
  if (t >= NPIX) return;
  int pw = t & 127, ph = (t >> 7) & 127, b = t >> 14;
  const _Float16* fp = feat + (size_t)t * 32;
  float s = bb[0];
  #pragma unroll
  for (int c = 0; c < 4; ++c) s += (float)fp[c] * w[c];
  float hidx = xy[b*2+0] * (float)(Hh-1), widx = xy[b*2+1] * (float)(Ww-1);
  int hI = (int)hidx, wI = (int)widx;
  int up = hI - AREA/2; up = up < 0 ? 0 : (up > HS-AREA ? HS-AREA : up);
  int lf = wI - AREA/2; lf = lf < 0 ? 0 : (lf > WS-AREA ? WS-AREA : lf);
  s += (ph >= up && ph < up + AREA && pw >= lf && pw < lf + AREA) ? 3.0f : -3.0f;
  float sal = xas[((size_t)b * (Aa*5) + 3) * (HS*WS) + ph*WS + pw];
  dmap[t] = sal / (1.0f + __expf(-s));
}

// ---------------------------------------------------------------------------
// Separable 17-tap Gaussian blur (sigma=8, 'reflect' padding)
// ---------------------------------------------------------------------------
__device__ __forceinline__ float gwk(int i) {
  float xs = (float)i - 8.0f;
  return __expf(-0.5f * (xs / 8.0f) * (xs / 8.0f));
}

__global__ void blur_h(const float* __restrict__ src, float* __restrict__ dst) {
  int t = blockIdx.x * blockDim.x + threadIdx.x;
  if (t >= NPIX) return;
  int pw = t & 127, ph = (t >> 7) & 127, b = t >> 14;
  const float* row = src + (size_t)b * (HS*WS) + (size_t)ph * WS;
  float s = 0.f, wn = 0.f;
  #pragma unroll
  for (int i = 0; i < 17; ++i) {
    int p = pw + i - 8;
    p = p < 0 ? -p : (p > 127 ? 254 - p : p);
    float wk = gwk(i); wn += wk;
    s += wk * row[p];
  }
  dst[t] = s / wn;
}

__global__ void blur_v(const float* __restrict__ src, float* __restrict__ dst) {
  int t = blockIdx.x * blockDim.x + threadIdx.x;
  if (t >= NPIX) return;
  int pw = t & 127, ph = (t >> 7) & 127, b = t >> 14;
  const float* img = src + (size_t)b * (HS*WS);
  float s = 0.f, wn = 0.f;
  #pragma unroll
  for (int i = 0; i < 17; ++i) {
    int p = ph + i - 8;
    p = p < 0 ? -p : (p > 127 ? 254 - p : p);
    float wk = gwk(i); wn += wk;
    s += wk * img[(size_t)p * WS + pw];
  }
  dst[t] = s / wn;
}

// ---------------------------------------------------------------------------
// Grid from density: 5x5 box of (edge-padded) blurred density times normalized
// padded coords, divided by (box + 1e-6). Output (B,HS,WS,2) in (x,y) order.
// ---------------------------------------------------------------------------
__global__ void grid_kernel(const float* __restrict__ dm, float* __restrict__ grid) {
  int t = blockIdx.x * blockDim.x + threadIdx.x;
  if (t >= NPIX) return;
  int j = t & 127, i = (t >> 7) & 127, b = t >> 14;
  const float* db = dm + (size_t)b * (HS*WS);
  float den = 0.f, sy = 0.f, sx = 0.f;
  #pragma unroll
  for (int u = 0; u < 5; ++u) {
    int ip = i + u - 2;
    float wy = 2.0f * (float)ip / 127.0f - 1.0f;
    int ic = ip < 0 ? 0 : (ip > 127 ? 127 : ip);
    #pragma unroll
    for (int v = 0; v < 5; ++v) {
      int jp = j + v - 2;
      float wx = 2.0f * (float)jp / 127.0f - 1.0f;
      int jc = jp < 0 ? 0 : (jp > 127 ? 127 : jp);
      float d = db[(size_t)ic * WS + jc];
      den += d; sy += d * wy; sx += d * wx;
    }
  }
  float id = 1.0f / (den + 1e-6f);
  grid[(size_t)t * 2 + 0] = sx * id;
  grid[(size_t)t * 2 + 1] = sy * id;
}

// ---------------------------------------------------------------------------
// Grid-sample all 4 crop levels (4 image ch bilinear from x, distance channel
// analytic), zeros padding, align_corners. Writes x_gs (f32, d_out) and
// normalized f16 NHWC32 activations for the seg CNN.
// ---------------------------------------------------------------------------
__global__ __launch_bounds__(256) void gsample(
    const float* __restrict__ x, const float* __restrict__ xy,
    const float* __restrict__ grid, float* __restrict__ xgs,
    _Float16* __restrict__ xn) {
  int t = blockIdx.x * blockDim.x + threadIdx.x;
  if (t >= NPIX) return;
  int j = t & 127, i = (t >> 7) & 127, b = t >> 14;

  float gx = grid[(size_t)t * 2 + 0];
  float gy = grid[(size_t)t * 2 + 1];
  float hidx = xy[b*2+0] * (float)(Hh-1), widx = xy[b*2+1] * (float)(Ww-1);
  int hI = (int)hidx, wI = (int)widx;
  float mh = fmaxf(hidx*hidx, (1023.f-hidx)*(1023.f-hidx));
  float mw = fmaxf(widx*widx, (1023.f-widx)*(1023.f-widx));
  float inv_maxd = rsqrtf(mh + mw);

  const float* xb  = x + (size_t)b * Cc * Hh * Ww;
  float*       xop = xgs + (size_t)b * Aa * 5 * HS * WS;
  _Float16*    xnp = xn + (size_t)t * 32;

  #pragma unroll
  for (int a = 0; a < 4; ++a) {
    const int chs = Hh >> a;
    int ua = hI - chs/2; ua = ua < 0 ? 0 : (ua > Hh-chs ? Hh-chs : ua);
    int la = wI - chs/2; la = la < 0 ? 0 : (la > Ww-chs ? Ww-chs : la);
    float fx = (gx + 1.0f) * 0.5f * (float)(chs - 1);
    float fy = (gy + 1.0f) * 0.5f * (float)(chs - 1);
    float x0f = floorf(fx), y0f = floorf(fy);
    float wx1 = fx - x0f, wy1 = fy - y0f;
    int xi0 = (int)x0f, yi0 = (int)y0f;

    float wgt[4] = {(1.f-wx1)*(1.f-wy1), wx1*(1.f-wy1), (1.f-wx1)*wy1, wx1*wy1};
    int   cx[4]  = {xi0, xi0+1, xi0,   xi0+1};
    int   cy[4]  = {yi0, yi0,   yi0+1, yi0+1};
    float acc[5] = {0.f, 0.f, 0.f, 0.f, 0.f};
    #pragma unroll
    for (int k = 0; k < 4; ++k) {
      bool valid = (cx[k] >= 0) && (cx[k] <= chs-1) && (cy[k] >= 0) && (cy[k] <= chs-1);
      int xcc = cx[k] < 0 ? 0 : (cx[k] > chs-1 ? chs-1 : cx[k]);
      int ycc = cy[k] < 0 ? 0 : (cy[k] > chs-1 ? chs-1 : cy[k]);
      float wv = valid ? wgt[k] : 0.0f;
      size_t off = (size_t)(ua + ycc) * Ww + (la + xcc);
      #pragma unroll
      for (int c = 0; c < 4; ++c) acc[c] += wv * xb[(size_t)c * Hh * Ww + off];
      float dh = (float)(ua + ycc) - hidx, dw = (float)(la + xcc) - widx;
      acc[4] += wv * sqrtf(dh*dh + dw*dw) * inv_maxd;
    }
    #pragma unroll
    for (int c = 0; c < 5; ++c) {
      xop[(a*5 + c) * (HS*WS) + i*WS + j] = acc[c];
      xnp[a*5 + c] = (_Float16)((acc[c] - c_mean[c]) / c_std[c]);
    }
  }
  #pragma unroll
  for (int c = 20; c < 32; ++c) xnp[c] = (_Float16)0.0f;
}

// ---------------------------------------------------------------------------
// seg head: 1x1 conv (32->1) + sigmoid, scaled by x_gs[:,0,3] -> y_pred
// ---------------------------------------------------------------------------
__global__ void final_head(const _Float16* __restrict__ feat,
                           const float* __restrict__ w, const float* __restrict__ bb,
                           const float* __restrict__ xgs, float* __restrict__ ypred) {
  int t = blockIdx.x * blockDim.x + threadIdx.x;
  if (t >= NPIX) return;
  int j = t & 127, i = (t >> 7) & 127, b = t >> 14;
  const _Float16* fp = feat + (size_t)t * 32;
  float s = bb[0];
  #pragma unroll
  for (int c = 0; c < 32; ++c) s += (float)fp[c] * w[c];
  float sal = xgs[((size_t)b * (Aa*5) + 3) * (HS*WS) + i*WS + j];
  ypred[t] = sal / (1.0f + __expf(-s));
}

// ---------------------------------------------------------------------------
extern "C" void kernel_launch(void* const* d_in, const int* in_sizes, int n_in,
                              void* d_out, int out_size, void* d_ws, size_t ws_size,
                              hipStream_t stream) {
  (void)in_sizes; (void)n_in; (void)out_size; (void)ws_size;
  const float* x  = (const float*)d_in[0];
  const float* xy = (const float*)d_in[1];
  // d_in[2..7]  : seg0  W0(4,20,3,3) b0 | W1(4,4,3,3) b1 | W2(1,4,1,1) b2
  // d_in[8..17] : seg   W0(32,20,3,3) b0 | W1..W3(32,32,3,3) b | W4(1,32,1,1) b4

  float* out    = (float*)d_out;
  float* y_pred = out;                       // 131072
  float* grid   = out + 131072;              // 262144
  float* dmap   = out + 131072 + 262144;     // 131072
  float* xas    = out + 524288;              // 2621440
  float* xgs    = out + 3145728;             // 2621440

  char* ws = (char*)d_ws;
  _Float16* act_in = (_Float16*)(ws);                     // 8 MB (xas_n, reused for xgs_n)
  _Float16* ping   = (_Float16*)(ws + (size_t)(8u  << 20));
  _Float16* pong   = (_Float16*)(ws + (size_t)(16u << 20));
  float*    blur1  = (float*)   (ws + (size_t)(24u << 20));
  float*    blur2  = (float*)   (ws + (size_t)(25u << 20));
  _Float16* wpk    = (_Float16*)(ws + (size_t)(26u << 20)); // 6 x 16384 halves
  float*    bpad   = (float*)   (ws + (size_t)(26u << 20) + 6 * 32768);

  // --- repack conv3x3 weights into WMMA B-fragment layout ---
  const int NPK = 9 * 2 * 32 * 16;
  const int lidx[6] = {2, 4, 8, 10, 12, 14};
  const int lcin[6] = {20, 4, 20, 32, 32, 32};
  const int lcou[6] = {4, 4, 32, 32, 32, 32};
  for (int l = 0; l < 6; ++l) {
    repack_w<<<(NPK + 255) / 256, 256, 0, stream>>>(
        (const float*)d_in[lidx[l]], wpk + (size_t)l * 16384, lcin[l], lcou[l]);
    pad_bias<<<1, 32, 0, stream>>>((const float*)d_in[lidx[l] + 1],
                                   bpad + l * 32, lcou[l]);
  }

  const int TPB = 256, GPX = (NPIX + TPB - 1) / TPB;
  build_xas<<<GPX, TPB, 0, stream>>>(x, xy, xas, act_in);

  // seg0 CNN (WMMA)
  const int CGRID = Bn * HS * 2;  // 2048 blocks of 128 threads (4 waves)
  conv3x3_wmma<<<CGRID, 128, 0, stream>>>(act_in, wpk + 0*16384, bpad + 0,   ping, 1);
  conv3x3_wmma<<<CGRID, 128, 0, stream>>>(ping,   wpk + 1*16384, bpad + 32,  pong, 1);
  dmap_head<<<GPX, TPB, 0, stream>>>(pong, (const float*)d_in[6],
                                     (const float*)d_in[7], xy, xas, dmap);

  blur_h<<<GPX, TPB, 0, stream>>>(dmap, blur1);
  blur_v<<<GPX, TPB, 0, stream>>>(blur1, blur2);
  grid_kernel<<<GPX, TPB, 0, stream>>>(blur2, grid);

  gsample<<<GPX, TPB, 0, stream>>>(x, xy, grid, xgs, act_in);  // act_in reused

  // seg CNN (WMMA)
  conv3x3_wmma<<<CGRID, 128, 0, stream>>>(act_in, wpk + 2*16384, bpad + 64,  ping, 1);
  conv3x3_wmma<<<CGRID, 128, 0, stream>>>(ping,   wpk + 3*16384, bpad + 96,  pong, 1);
  conv3x3_wmma<<<CGRID, 128, 0, stream>>>(pong,   wpk + 4*16384, bpad + 128, ping, 1);
  conv3x3_wmma<<<CGRID, 128, 0, stream>>>(ping,   wpk + 5*16384, bpad + 160, pong, 1);

  final_head<<<GPX, TPB, 0, stream>>>(pong, (const float*)d_in[16],
                                      (const float*)d_in[17], xgs, y_pred);
}
